// BiLSTM_83219286327400
// MI455X (gfx1250) — matrix operand order repaired and checked by reading, not compiled, not run on hardware
//
#include <hip/hip_runtime.h>
#include <hip/hip_bf16.h>

// ---------------------------------------------------------------------------
// BiLSTM (T=1024, B=128, H=256, E=20, C=20) for gfx1250 / MI455X.
// Persistent-scan design, v6:
//   - 64 WGs x 512 threads: 32 per direction, each owns 32 gate columns
//     (8 hidden units). Wave = (mtile 0..7) x (jtile 0..1) -> ONE 16x16 tile.
//   - Recurrent weights live in VGPRs for the whole 1024-step scan (72 regs).
//   - All 9 A-fragments preloaded in one burst (18 b128 loads, distinct regs,
//     pinned with sched_barrier) -> incremental s_wait_loadcnt, one L2
//     latency per step, no spills (~175 VGPRs/wave).
//   - Gate pre-activations bounce through a 16 KB LDS tile; each thread owns
//     2 (batch,unit) updates read as one float4 (i,f,g,o adjacent).
//   - Activations use v_exp_f32 + v_rcp_f32 (no IEEE division sequences).
//   - Input projection fused into the recurrent GEMM (K = 256+32pad = 288).
//   - Max-pool over T folded into the scan as per-thread running max.
//   - One device-scope barrier per step per direction (monotonic counter).
// ---------------------------------------------------------------------------

typedef __attribute__((ext_vector_type(16))) __bf16 v16bf;
typedef __attribute__((ext_vector_type(8)))  float  v8f;

#define T_STEPS 1024
#define BATCH   128
#define HID     256
#define EMB     20
#define NGATE   1024      // 4*HID
#define KPAD    288       // HID + 32 (EMB padded to 32)
#define KT      9         // KPAD / 32
#define NWG_DIR 32        // workgroups per direction
#define NCOLS   32        // gate columns per workgroup (8 hidden units)
#define UNITS   8         // hidden units per workgroup
#define CLS     20
#define THREADS 512       // 16 waves: wave = (mtile 0..7) x (jtile 0..1)

union FragU {
    uint4 u4[2];
    v16bf v;
};

__device__ __forceinline__ unsigned short f2bf(float f) {
    __hip_bfloat16 h = __float2bfloat16(f);
    return *reinterpret_cast<unsigned short*>(&h);
}

__device__ __forceinline__ float frcp(float x) {
    return __builtin_amdgcn_rcpf(x);                 // v_rcp_f32
}
__device__ __forceinline__ float fsigmoid(float x) {
    return frcp(1.f + __expf(-x));                   // 1 exp + 1 rcp
}
__device__ __forceinline__ float ftanh_f(float x) {
    x = fminf(fmaxf(x, -15.f), 15.f);
    return 1.f - 2.f * frcp(__expf(2.f * x) + 1.f);  // 1 exp + 1 rcp
}

// ---------------------------------------------------------------------------
// Build W_cat[dir][n][k] (bf16, row-major, rowlen KPAD).
// Gate rows reordered: n = 4*u + gate (gate order i,f,g,o), src row r = gate*H+u.
// k < 256 -> Whh[r][k]; 256 <= k < 276 -> Wih[r][k-256]; else 0.
// ---------------------------------------------------------------------------
__global__ __launch_bounds__(256)
void prep_wcat(const float* __restrict__ Wih_f, const float* __restrict__ Whh_f,
               const float* __restrict__ Wih_b, const float* __restrict__ Whh_b,
               unsigned short* __restrict__ Wcat) {
    int idx = blockIdx.x * 256 + threadIdx.x;
    const int total = 2 * NGATE * KPAD;
    if (idx >= total) return;
    int dir = idx / (NGATE * KPAD);
    int rem = idx - dir * (NGATE * KPAD);
    int n = rem / KPAD;
    int k = rem - n * KPAD;
    int gi = n & 3, u = n >> 2;
    int r = gi * HID + u;
    const float* Wih = dir ? Wih_b : Wih_f;
    const float* Whh = dir ? Whh_b : Whh_f;
    float v = 0.f;
    if (k < HID)             v = Whh[r * HID + k];
    else if (k < HID + EMB)  v = Wih[r * EMB + (k - HID)];
    Wcat[idx] = f2bf(v);
}

// ---------------------------------------------------------------------------
// Init A-buffer parity 0 (h0 bf16 + gathered xe for the first step of each
// direction), reordered bias, and barrier counters.
// ---------------------------------------------------------------------------
__global__ __launch_bounds__(256)
void prep_misc(const int* __restrict__ x, const float* __restrict__ h0,
               const float* __restrict__ emb,
               const float* __restrict__ b_f, const float* __restrict__ b_b,
               unsigned short* __restrict__ Abuf, float* __restrict__ bcat,
               unsigned* __restrict__ bar) {
    int i = blockIdx.x * 256 + threadIdx.x;
    if (i < 2 * BATCH * HID) {                       // h0 -> A[dir][parity0] cols 0..255
        int dir = i / (BATCH * HID);
        int rem = i - dir * (BATCH * HID);
        int b = rem / HID;
        int u = rem - b * HID;
        Abuf[(size_t)(dir * 2) * BATCH * KPAD + (size_t)b * KPAD + u] = f2bf(h0[i]);
        return;
    }
    i -= 2 * BATCH * HID;
    if (i < 2 * BATCH * 32) {                        // xe_0 -> cols 256..287 (pad zeros)
        int dir = i / (BATCH * 32);
        int rem = i - dir * (BATCH * 32);
        int b = rem >> 5, e = rem & 31;
        int t0 = dir ? (T_STEPS - 1) : 0;
        float v = (e < EMB) ? emb[(size_t)x[t0 * BATCH + b] * EMB + e] : 0.f;
        Abuf[(size_t)(dir * 2) * BATCH * KPAD + (size_t)b * KPAD + HID + e] = f2bf(v);
        return;
    }
    i -= 2 * BATCH * 32;
    if (i < 2 * NGATE) {                             // reordered bias
        int dir = i / NGATE;
        int n = i - dir * NGATE;
        int r = (n & 3) * HID + (n >> 2);
        bcat[i] = (dir ? b_b : b_f)[r];
        return;
    }
    i -= 2 * NGATE;
    if (i < 64) bar[i] = 0;                          // barrier counters
}

// ---------------------------------------------------------------------------
// Persistent recurrent scan. Grid = 64 WGs x 512 threads.
//   blockIdx: dir = bi>>5, nb = bi&31 -> gate columns [32*nb, 32*nb+32)
//   wave w: jtile = w&1 (16-col strip), mtile = w>>1 (16-row M tile).
// ---------------------------------------------------------------------------
__global__ __launch_bounds__(THREADS)
void lstm_scan(const int* __restrict__ x,
               const float* __restrict__ c0,
               const float* __restrict__ emb,
               const unsigned short* __restrict__ Wcat,
               const float* __restrict__ bcat,
               unsigned short* __restrict__ Abuf,
               float* __restrict__ pooled,
               unsigned* __restrict__ bar,
               float* __restrict__ out_hn,
               float* __restrict__ out_cn) {
    __shared__ float g_lds[BATCH * NCOLS];           // 16 KB gate staging

    const int tid  = threadIdx.x;
    const int lane = tid & 31;
    const int wave = tid >> 5;
    const int dir  = blockIdx.x >> 5;
    const int nb   = blockIdx.x & 31;
    const int n0   = nb * NCOLS;

    const int jtile = wave & 1;                      // N strip (16 cols)
    const int mtile = wave >> 1;                     // M tile (rows mtile*16..+15)
    const int lmod  = lane & 15;
    const int mhi   = lane >> 4;

    // ---- preload this wave's B fragments into registers (all 1024 steps) ----
    FragU wf[KT];
    {
        const unsigned short* wp =
            Wcat + ((size_t)dir * NGATE + n0 + jtile * 16 + lmod) * KPAD + mhi * 16;
#pragma unroll
        for (int kt = 0; kt < KT; ++kt) {
            wf[kt].u4[0] = *(const uint4*)(wp + kt * 32);
            wf[kt].u4[1] = *(const uint4*)(wp + kt * 32 + 8);
        }
    }
    const float bias = bcat[dir * NGATE + n0 + jtile * 16 + lmod];

    // ---- per-thread cell state: 2 (batch,unit) pairs ----
    const int Mown   = tid >> 2;
    const int ulbase = (tid & 3) * 2;
    float creg[2], hmax[2];
#pragma unroll
    for (int i = 0; i < 2; ++i) {
        creg[i] = c0[(size_t)dir * BATCH * HID + (size_t)Mown * HID + nb * UNITS + ulbase + i];
        hmax[i] = -3.0e38f;
    }

    const int mrow = mtile * 16 + lmod;              // A row this lane loads

    unsigned* cnt = bar + dir * 16;

    for (int t = 0; t < T_STEPS; ++t) {
        const unsigned short* Acur =
            Abuf + (size_t)(dir * 2 + (t & 1)) * (BATCH * KPAD);
        unsigned short* Anxt =
            Abuf + (size_t)(dir * 2 + ((t & 1) ^ 1)) * (BATCH * KPAD);

        // ---- preload ALL 9 A fragments (18 b128 loads, distinct VGPRs) ----
        FragU af[KT];
        {
            const unsigned short* ar = Acur + (size_t)mrow * KPAD + mhi * 16;
#pragma unroll
            for (int kt = 0; kt < KT; ++kt) {
                af[kt].u4[0] = *(const uint4*)(ar + kt * 32);
                af[kt].u4[1] = *(const uint4*)(ar + kt * 32 + 8);
            }
        }
        // Pin: all loads issue before any WMMA (one L2 latency, not nine).
        __builtin_amdgcn_sched_barrier(0);

        v8f acc;
#pragma unroll
        for (int e = 0; e < 8; ++e) acc[e] = 0.f;

        // ---- g = [h_prev | xe_t] @ Wstrip^T : 9 WMMAs, operands in VGPRs ----
#pragma unroll
        for (int kt = 0; kt < KT; ++kt) {
            acc = __builtin_amdgcn_wmma_f32_16x16x32_bf16(
                false, af[kt].v, false, wf[kt].v, (short)0, acc, false, false);
        }
        __builtin_amdgcn_sched_barrier(0);

        // ---- stage g (+bias) to LDS in [M][32] layout ----
        {
            const int col = jtile * 16 + lmod;
#pragma unroll
            for (int r = 0; r < 8; ++r) {
                g_lds[(mtile * 16 + r + 8 * mhi) * NCOLS + col] = acc[r] + bias;
            }
        }
        __syncthreads();

        // ---- per-thread gate update: 2 units, each one float4 (i,f,g,o) ----
#pragma unroll
        for (int i = 0; i < 2; ++i) {
            int ul = ulbase + i;
            float4 g4 = *(const float4*)&g_lds[Mown * NCOLS + ul * 4];
            float iv = fsigmoid(g4.x);
            float fv = fsigmoid(g4.y);
            float gv = ftanh_f(g4.z);
            float ov = fsigmoid(g4.w);
            float cv = fv * creg[i] + iv * gv;
            creg[i] = cv;
            float hv = ov * ftanh_f(cv);
            hmax[i] = fmaxf(hmax[i], hv);
            int u = nb * UNITS + ul;
            Anxt[(size_t)Mown * KPAD + u] = f2bf(hv);
            if (t == T_STEPS - 1) {
                out_hn[(size_t)dir * BATCH * HID + (size_t)Mown * HID + u] = hv;
                out_cn[(size_t)dir * BATCH * HID + (size_t)Mown * HID + u] = cv;
            }
        }

        // ---- WG0 of each direction gathers xe_{t+1} into the next buffer ----
        if (nb == 0 && t + 1 < T_STEPS) {
            int xi = dir ? (T_STEPS - 2 - t) : (t + 1);
            for (int q = tid; q < BATCH * 32; q += THREADS) {
                int b = q >> 5, e = q & 31;
                float v = (e < EMB) ? emb[(size_t)x[xi * BATCH + b] * EMB + e] : 0.f;
                Anxt[(size_t)b * KPAD + HID + e] = f2bf(v);
            }
        }

        // ---- per-direction device barrier (monotonic counter) ----
        __threadfence();
        __syncthreads();
        if (tid == 0) {
            __hip_atomic_fetch_add(cnt, 1u, __ATOMIC_ACQ_REL, __HIP_MEMORY_SCOPE_AGENT);
            unsigned want = (unsigned)(NWG_DIR * (t + 1));
            while (__hip_atomic_load(cnt, __ATOMIC_ACQUIRE, __HIP_MEMORY_SCOPE_AGENT) < want)
                __builtin_amdgcn_s_sleep(1);
        }
        __syncthreads();

        if (t + 1 < T_STEPS) {
            const unsigned short* an =
                Abuf + (size_t)(dir * 2 + ((t & 1) ^ 1)) * (BATCH * KPAD) +
                (size_t)mrow * KPAD;
            __builtin_prefetch(an, 0, 1);
        }
    }

    // ---- write pooled max (B x 2H): forward units first, then backward ----
#pragma unroll
    for (int i = 0; i < 2; ++i) {
        int u = nb * UNITS + ulbase + i;
        pooled[(size_t)Mown * (2 * HID) + dir * HID + u] = hmax[i];
    }
}

// ---------------------------------------------------------------------------
// Output head: logits = pooled @ W_out^T + b_out, softmax. One block, 128 thr.
// ---------------------------------------------------------------------------
__global__ __launch_bounds__(128)
void head_kernel(const float* __restrict__ pooled,
                 const float* __restrict__ W_out,
                 const float* __restrict__ b_out,
                 float* __restrict__ probs) {
    int b = threadIdx.x;
    float acc[CLS];
#pragma unroll
    for (int c = 0; c < CLS; ++c) acc[c] = b_out[c];
    const float* p = pooled + (size_t)b * (2 * HID);
    for (int k = 0; k < 2 * HID; ++k) {
        float pv = p[k];
#pragma unroll
        for (int c = 0; c < CLS; ++c)
            acc[c] += pv * W_out[(size_t)c * (2 * HID) + k];
    }
    float m = acc[0];
#pragma unroll
    for (int c = 1; c < CLS; ++c) m = fmaxf(m, acc[c]);
    float s = 0.f;
#pragma unroll
    for (int c = 0; c < CLS; ++c) { acc[c] = __expf(acc[c] - m); s += acc[c]; }
    float inv = frcp(s);
#pragma unroll
    for (int c = 0; c < CLS; ++c) probs[(size_t)b * CLS + c] = acc[c] * inv;
}

// ---------------------------------------------------------------------------
extern "C" void kernel_launch(void* const* d_in, const int* in_sizes, int n_in,
                              void* d_out, int out_size, void* d_ws, size_t ws_size,
                              hipStream_t stream) {
    const int*   x     = (const int*)  d_in[0];
    const float* h0    = (const float*)d_in[1];
    const float* c0    = (const float*)d_in[2];
    const float* emb   = (const float*)d_in[3];
    const float* Wih_f = (const float*)d_in[4];
    const float* Whh_f = (const float*)d_in[5];
    const float* b_f   = (const float*)d_in[6];
    const float* Wih_b = (const float*)d_in[7];
    const float* Whh_b = (const float*)d_in[8];
    const float* b_b   = (const float*)d_in[9];
    const float* W_out = (const float*)d_in[10];
    const float* b_out = (const float*)d_in[11];

    char* ws = (char*)d_ws;
    unsigned short* Wcat = (unsigned short*)(ws);            // 2*1024*288*2 = 1179648 B
    float*          bcat = (float*)(ws + 1179648);           // 8192 B
    unsigned short* Abuf = (unsigned short*)(ws + 1187840);  // 2*2*128*288*2 = 294912 B
    float*        pooled = (float*)(ws + 1482752);           // 128*512*4 = 262144 B
    unsigned*        bar = (unsigned*)(ws + 1744896);        // 256 B

    float* probs  = (float*)d_out;                           // 128*20
    float* out_hn = probs + BATCH * CLS;                     // 2*128*256
    float* out_cn = out_hn + 2 * BATCH * HID;                // 2*128*256

    prep_wcat<<<(2 * NGATE * KPAD + 255) / 256, 256, 0, stream>>>(
        Wih_f, Whh_f, Wih_b, Whh_b, Wcat);

    const int misc_total = 2 * BATCH * HID + 2 * BATCH * 32 + 2 * NGATE + 64;
    prep_misc<<<(misc_total + 255) / 256, 256, 0, stream>>>(
        x, h0, emb, b_f, b_b, Abuf, bcat, bar);

    lstm_scan<<<2 * NWG_DIR, THREADS, 0, stream>>>(
        x, c0, emb, Wcat, bcat, Abuf, pooled, bar, out_hn, out_cn);

    head_kernel<<<1, 128, 0, stream>>>(pooled, W_out, b_out, probs);
}
